// GraphAggr_57329223467063
// MI455X (gfx1250) — matrix operand order repaired and checked by reading, not compiled, not run on hardware
//
#include <hip/hip_runtime.h>
#include <hip/hip_bf16.h>

typedef __attribute__((ext_vector_type(16))) _Float16 v16h;
typedef __attribute__((ext_vector_type(8)))  _Float16 v8h;
typedef __attribute__((ext_vector_type(8)))  float    v8f;

#define DD 128
#define STRIDE 136            // f16 row stride in LDS (pad 8 -> keeps 16B alignment, skews banks)
#define ROWS_PER_BLOCK 128

union Frag { v16h v; v8h h[2]; };

// ---------------------------------------------------------------------------
// Pass 1: gate[n] = prelu(prelu(x@W1+b1)@W2+b2) @ W3 + b3, via WMMA f16->f32
// ---------------------------------------------------------------------------
__global__ __launch_bounds__(256) void gate_kernel(
    const float* __restrict__ x,
    const float* __restrict__ W1, const float* __restrict__ b1, const float* __restrict__ a1,
    const float* __restrict__ W2, const float* __restrict__ b2, const float* __restrict__ a2,
    const float* __restrict__ W3, const float* __restrict__ b3,
    float* __restrict__ gate, int N)
{
    __shared__ _Float16 sX[ROWS_PER_BLOCK * STRIDE];  // x tile, f16, row-major
    __shared__ _Float16 sW[DD * STRIDE];              // weights TRANSPOSED: sW[n*STRIDE + k]
    __shared__ _Float16 sH[ROWS_PER_BLOCK * STRIDE];  // hidden activations, f16

    const int t    = threadIdx.x;
    const int lane = t & 31;
    const int wave = t >> 5;          // 8 waves
    const int half = lane >> 4;       // 0 or 1
    const int l16  = lane & 15;
    const int row0  = blockIdx.x * ROWS_PER_BLOCK;
    const int nrows = min(ROWS_PER_BLOCK, N - row0);
    const int Mbase = wave * 16;

    // ---- stage x tile (f32 -> f16), zero-pad ragged tail ----
    for (int i = t; i < ROWS_PER_BLOCK * DD; i += 256) {
        int r = i >> 7, c = i & 127;
        float v = (r < nrows) ? x[(size_t)(row0 + r) * DD + c] : 0.0f;
        sX[r * STRIDE + c] = (_Float16)v;
    }
    // ---- stage W1 transposed (f32 -> f16) ----
    for (int i = t; i < DD * DD; i += 256) {
        int k = i >> 7, n = i & 127;
        sW[n * STRIDE + k] = (_Float16)W1[k * DD + n];
    }
    __syncthreads();

    const float alpha1 = a1[0];
    const float alpha2 = a2[0];

    // ---- A fragments from sX (hoisted over N tiles) ----
    // ISA 16-bit A layout: lane M = lane%16; VGPR0-3 = K {8h..8h+7}; VGPR4-7 = K {16+8h..}
    Frag A[4];
    {
        const _Float16* rowp = sX + (Mbase + l16) * STRIDE;
        for (int kc = 0; kc < 4; ++kc) {
            const _Float16* p = rowp + kc * 32 + half * 8;
            A[kc].h[0] = *(const v8h*)p;          // ds_load_b128
            A[kc].h[1] = *(const v8h*)(p + 16);   // ds_load_b128
        }
    }

    // ---- GEMM1: h1 = prelu(x @ W1 + b1) -> sH (f16) ----
    for (int nt = 0; nt < 8; ++nt) {
        v8f acc = {};
        const int n = nt * 16 + l16;
        const _Float16* colp = sW + n * STRIDE;   // transposed: contiguous in K
        for (int kc = 0; kc < 4; ++kc) {
            // ISA 16-bit B layout: lane N = lane%16; VGPR0-3 = K {8h..8h+7}; VGPR4-7 = K {16+8h..}
            Frag B;
            const _Float16* q = colp + kc * 32 + half * 8;
            B.h[0] = *(const v8h*)q;
            B.h[1] = *(const v8h*)(q + 16);
            acc = __builtin_amdgcn_wmma_f32_16x16x32_f16(
                false, A[kc].v, false, B.v, (short)0, acc, false, false);
        }
        const float bias = b1[n];
        for (int r = 0; r < 8; ++r) {             // C/D: elem r -> M = r + 8*half, N = l16
            float v = acc[r] + bias;
            v = (v >= 0.0f) ? v : alpha1 * v;
            sH[(Mbase + r + 8 * half) * STRIDE + n] = (_Float16)v;
        }
    }

    __syncthreads();
    // ---- restage W2 transposed ----
    for (int i = t; i < DD * DD; i += 256) {
        int k = i >> 7, n = i & 127;
        sW[n * STRIDE + k] = (_Float16)W2[k * DD + n];
    }
    __syncthreads();

    // ---- A fragments from sH (wave reads its own 16 rows) ----
    Frag A2[4];
    {
        const _Float16* rowp = sH + (Mbase + l16) * STRIDE;
        for (int kc = 0; kc < 4; ++kc) {
            const _Float16* p = rowp + kc * 32 + half * 8;
            A2[kc].h[0] = *(const v8h*)p;
            A2[kc].h[1] = *(const v8h*)(p + 16);
        }
    }

    // ---- GEMM2: h2 = prelu(h1 @ W2 + b2), kept in registers ----
    v8f acc2[8];
    for (int nt = 0; nt < 8; ++nt) {
        v8f acc = {};
        const int n = nt * 16 + l16;
        const _Float16* colp = sW + n * STRIDE;
        for (int kc = 0; kc < 4; ++kc) {
            Frag B;
            const _Float16* q = colp + kc * 32 + half * 8;
            B.h[0] = *(const v8h*)q;
            B.h[1] = *(const v8h*)(q + 16);
            acc = __builtin_amdgcn_wmma_f32_16x16x32_f16(
                false, A2[kc].v, false, B.v, (short)0, acc, false, false);
        }
        const float bias = b2[n];
        for (int r = 0; r < 8; ++r) {
            float v = acc[r] + bias;
            acc2[nt][r] = (v >= 0.0f) ? v : alpha2 * v;
        }
    }

    // ---- gate = h2 @ W3 + b3 : per-VGPR partial + 16-lane butterfly reduce ----
    float w3v[8];
    for (int nt = 0; nt < 8; ++nt) w3v[nt] = W3[nt * 16 + l16];
    const float b3v = b3[0];
    for (int r = 0; r < 8; ++r) {
        float p = 0.0f;
        for (int nt = 0; nt < 8; ++nt) p += acc2[nt][r] * w3v[nt];
        for (int off = 8; off >= 1; off >>= 1) p += __shfl_xor(p, off, 16);
        const int m = Mbase + r + 8 * half;       // row index of this partial
        if (l16 == 0 && m < nrows) gate[row0 + m] = p + b3v;
    }
}

// ---------------------------------------------------------------------------
// Pass 2: single-block softmax stats over gate[] -> stats = {max, sum_exp}
// ---------------------------------------------------------------------------
__global__ __launch_bounds__(1024) void softmax_stats_kernel(
    const float* __restrict__ gate, int N, float* __restrict__ stats)
{
    __shared__ float red[1024];
    const int t = threadIdx.x;
    float m = -3.402823e38f;
    for (int i = t; i < N; i += 1024) m = fmaxf(m, gate[i]);
    red[t] = m; __syncthreads();
    for (int s = 512; s >= 1; s >>= 1) {
        if (t < s) red[t] = fmaxf(red[t], red[t + s]);
        __syncthreads();
    }
    const float gmax = red[0];
    __syncthreads();
    float z = 0.0f;
    for (int i = t; i < N; i += 1024) z += __expf(gate[i] - gmax);
    red[t] = z; __syncthreads();
    for (int s = 512; s >= 1; s >>= 1) {
        if (t < s) red[t] += red[t + s];
        __syncthreads();
    }
    if (t == 0) { stats[0] = gmax; stats[1] = red[0]; }
}

__global__ void zero_out_kernel(float* __restrict__ out) { out[threadIdx.x] = 0.0f; }

// ---------------------------------------------------------------------------
// Pass 3: out[d] = sum_n softmax(gate)[n] * x[n][d]   (fp32 global atomics)
// ---------------------------------------------------------------------------
__global__ __launch_bounds__(256) void weighted_sum_kernel(
    const float* __restrict__ x, const float* __restrict__ gate,
    const float* __restrict__ stats, float* __restrict__ out,
    int N, int rowsPerBlock)
{
    __shared__ float partial[DD];
    const float gmax = stats[0];
    const float invZ = 1.0f / stats[1];
    const int t  = threadIdx.x;
    const int d  = t & 127;
    const int rg = t >> 7;                        // 0 or 1
    const int base = blockIdx.x * rowsPerBlock;
    float acc = 0.0f;
    for (int r = rg; r < rowsPerBlock; r += 2) {
        const int n = base + r;
        if (n < N) acc += __expf(gate[n] - gmax) * x[(size_t)n * DD + d];
    }
    if (rg == 1) partial[d] = acc;
    __syncthreads();
    if (rg == 0) atomicAdd(&out[d], (acc + partial[d]) * invZ);
}

extern "C" void kernel_launch(void* const* d_in, const int* in_sizes, int n_in,
                              void* d_out, int out_size, void* d_ws, size_t ws_size,
                              hipStream_t stream)
{
    const float* x  = (const float*)d_in[0];
    const float* W1 = (const float*)d_in[1];
    const float* b1 = (const float*)d_in[2];
    const float* a1 = (const float*)d_in[3];
    const float* W2 = (const float*)d_in[4];
    const float* b2 = (const float*)d_in[5];
    const float* a2 = (const float*)d_in[6];
    const float* W3 = (const float*)d_in[7];
    const float* b3 = (const float*)d_in[8];
    float* out = (float*)d_out;
    const int N = in_sizes[0] / DD;

    float* gate  = (float*)d_ws;      // N floats
    float* stats = gate + N;          // {max, Z}

    const int nBlocks = (N + ROWS_PER_BLOCK - 1) / ROWS_PER_BLOCK;
    gate_kernel<<<nBlocks, 256, 0, stream>>>(x, W1, b1, a1, W2, b2, a2, W3, b3, gate, N);
    softmax_stats_kernel<<<1, 1024, 0, stream>>>(gate, N, stats);
    zero_out_kernel<<<1, DD, 0, stream>>>(out);
    const int rowsPerBlock = 512;
    weighted_sum_kernel<<<(N + rowsPerBlock - 1) / rowsPerBlock, 256, 0, stream>>>(
        x, gate, stats, out, N, rowsPerBlock);
}
